// AgentEmbeddingLayer_43954695307298
// MI455X (gfx1250) — matrix operand order repaired and checked by reading, not compiled, not run on hardware
//
#include <hip/hip_runtime.h>
#include <hip/hip_bf16.h>

typedef __attribute__((ext_vector_type(16))) _Float16 v16h;
typedef __attribute__((ext_vector_type(8)))  _Float16 h8;
typedef __attribute__((ext_vector_type(8)))  float    v8f;

// ---------------------------------------------------------------------------
// Stage a weight matrix W (f32, K x N row-major) into WMMA-swizzled f16 layout:
//   Bs[(((tn*(K/32)) + kc)*32 + lane)*16 + j] = W[(kc*32 + 16*(lane>>4) + j)*N + tn*16 + (lane&15)]
// so that in the GEMM each lane reads its 16 B halfs as one contiguous 32B run.
// ---------------------------------------------------------------------------
__global__ void k_stage_B(const float* __restrict__ W, _Float16* __restrict__ Bs, int K, int N)
{
    int total = K * N;
    int idx = blockIdx.x * blockDim.x + threadIdx.x;
    if (idx >= total) return;
    int j    = idx & 15;
    int lane = (idx >> 4) & 31;
    int rest = idx >> 9;
    int KC   = K >> 5;
    int kc   = rest % KC;
    int tn   = rest / KC;
    int krow = kc * 32 + ((lane >> 4) << 4) + j;
    int col  = tn * 16 + (lane & 15);
    Bs[idx] = (_Float16)W[(size_t)krow * N + col];
}

// ---------------------------------------------------------------------------
// Build level-0 RPE: features (dist, rel-angle, rel-heading, rel-idx) @ rt_W + rt_b
// x: (192,5,64) f32.  Output rpe f32 + f16, shape (192,64,64,32) row-major.
// ---------------------------------------------------------------------------
__global__ void k_build_rpe(const float* __restrict__ x, const float* __restrict__ Wt,
                            const float* __restrict__ bt, float* __restrict__ rpe,
                            _Float16* __restrict__ rpeh)
{
    int idx = blockIdx.x * blockDim.x + threadIdx.x;
    const int total = 192 * 64 * 64;
    if (idx >= total) return;
    int j = idx & 63;
    int i = (idx >> 6) & 63;
    int b = idx >> 12;
    const float* xb = x + (size_t)b * 5 * 64;
    float pix = xb[0 * 64 + i], piy = xb[1 * 64 + i];
    float pjx = xb[0 * 64 + j], pjy = xb[1 * 64 + j];
    float hi  = xb[4 * 64 + i], hj  = xb[4 * 64 + j];
    float rx = pix - pjx, ry = piy - pjy;
    float sq = rx * rx + ry * ry;
    float r0 = (sq > 0.f) ? sqrtf(sq) : 0.f;
    float cx = cosf(hi) - cosf(hj);
    float cy = sinf(hi) - sinf(hj);
    float cross = cx * ry - cy * rx;
    float dot   = cx * rx + cy * ry;
    float r1 = (cross == 0.f && dot == 0.f) ? 0.f : atan2f(cross, dot);
    float dhd = hi - hj;
    float r2 = atan2f(sinf(dhd), cosf(dhd));
    float r3 = (float)(i - j);
    size_t base = (size_t)idx * 32;
#pragma unroll
    for (int e = 0; e < 32; ++e) {
        float vv = r0 * Wt[e] + r1 * Wt[32 + e] + r2 * Wt[64 + e] + r3 * Wt[96 + e] + bt[e];
        rpe[base + e]  = vv;
        rpeh[base + e] = (_Float16)vv;
    }
}

// ---------------------------------------------------------------------------
// Embed conv1d: x[:, :4, :] (192,4,64) * w(32,4,3) + b -> h (192,64,32) channel-last
// ---------------------------------------------------------------------------
__global__ void k_embed(const float* __restrict__ x, const float* __restrict__ w,
                        const float* __restrict__ bias, float* __restrict__ h)
{
    int idx = blockIdx.x * blockDim.x + threadIdx.x;
    const int total = 192 * 64 * 32;
    if (idx >= total) return;
    int o = idx & 31;
    int t = (idx >> 5) & 63;
    int b = idx >> 11;
    float s = bias[o];
#pragma unroll
    for (int c = 0; c < 4; ++c)
#pragma unroll
        for (int kk = 0; kk < 3; ++kk) {
            int tt = t + kk - 1;
            if (tt >= 0 && tt < 64)
                s += x[(size_t)b * 320 + c * 64 + tt] * w[(o * 4 + c) * 3 + kk];
        }
    h[((size_t)b * 64 + t) * 32 + o] = s;
}

// ---------------------------------------------------------------------------
// Row LayerNorm: one wave32 per row, shuffle reductions. y / yh nullable.
// ---------------------------------------------------------------------------
__global__ void k_layernorm(const float* __restrict__ x, const float* __restrict__ g,
                            const float* __restrict__ b, float* __restrict__ y,
                            _Float16* __restrict__ yh, int M, int D)
{
    int row  = blockIdx.x;
    int lane = threadIdx.x;
    const float* xr = x + (size_t)row * D;
    float s = 0.f;
    for (int c = lane; c < D; c += 32) s += xr[c];
#pragma unroll
    for (int o = 16; o; o >>= 1) s += __shfl_xor(s, o, 32);
    float mean = s / (float)D;
    float vs = 0.f;
    for (int c = lane; c < D; c += 32) { float d = xr[c] - mean; vs += d * d; }
#pragma unroll
    for (int o = 16; o; o >>= 1) vs += __shfl_xor(vs, o, 32);
    float inv = rsqrtf(vs / (float)D + 1e-5f);
    for (int c = lane; c < D; c += 32) {
        float val = (xr[c] - mean) * inv * g[c] + b[c];
        if (y)  y[(size_t)row * D + c]  = val;
        if (yh) yh[(size_t)row * D + c] = (_Float16)val;
    }
}

// ---------------------------------------------------------------------------
// WMMA GEMM: C[M,N] = act(A[M,K](f16,rm) x Bs[K,N](f16, pre-swizzled) + bias) + addend
// One wave32 per 16x16 C tile via v_wmma_f32_16x16x32_f16; 4 waves per block.
// A: two contiguous 8-half (b128) loads per k-step in contiguous k order (k = 16*(lane>>4) + s), same as Bs.
// B: one contiguous 16-half (2xb128) load per lane from the swizzled panel.
// ---------------------------------------------------------------------------
__global__ void k_gemm_f16(const _Float16* __restrict__ A, const _Float16* __restrict__ Bs,
                           const float* __restrict__ bias, const float* __restrict__ addend,
                           float* __restrict__ C, _Float16* __restrict__ Ch,
                           int M, int N, int K, int act)
{
    int tileM = blockIdx.x * 4 + threadIdx.y;
    int tileN = blockIdx.y;
    int lane  = threadIdx.x;
    int mlane = lane & 15;
    int g     = lane >> 4;
    int m0 = tileM * 16;
    int n0 = tileN * 16;
    int KC = K >> 5;
    v8f acc = {};
    const _Float16* Arow   = A + (size_t)(m0 + mlane) * K + 16 * g;
    const _Float16* Bpanel = Bs + (((size_t)tileN * KC) * 32 + lane) * 16;
    for (int kc = 0; kc < KC; ++kc) {
        const _Float16* Ap = Arow + kc * 32;
        h8 alo = *(const h8*)Ap;
        h8 ahi = *(const h8*)(Ap + 8);
        __builtin_prefetch(Ap + 32, 0, 0);  // speculative: next A k-chunk
        v16h av = __builtin_shufflevector(alo, ahi, 0, 1, 2, 3, 4, 5, 6, 7,
                                          8, 9, 10, 11, 12, 13, 14, 15);
        const _Float16* Bp = Bpanel + (size_t)kc * 512;
        h8 blo = *(const h8*)Bp;
        h8 bhi = *(const h8*)(Bp + 8);
        v16h bv = __builtin_shufflevector(blo, bhi, 0, 1, 2, 3, 4, 5, 6, 7,
                                          8, 9, 10, 11, 12, 13, 14, 15);
        acc = __builtin_amdgcn_wmma_f32_16x16x32_f16(false, av, false, bv,
                                                     (short)0, acc, false, false);
    }
    float bcol = bias ? bias[n0 + mlane] : 0.f;
#pragma unroll
    for (int r = 0; r < 8; ++r) {
        int row = m0 + r + 8 * g;
        size_t idx = (size_t)row * N + (size_t)(n0 + mlane);
        float vv = acc[r] + bcol;
        if (act == 1)  // exact GELU
            vv = 0.5f * vv * (1.0f + erff(vv * 0.70710678118f));
        if (addend) vv += addend[idx];
        if (C)  C[idx]  = vv;
        if (Ch) Ch[idx] = (_Float16)vv;
    }
    (void)M;
}

// ---------------------------------------------------------------------------
// Attention: one wave32 per (b, h, i).  dh = 16 at every level.
// logits = (q.k + q.rk) * 0.25 ; softmax ; o = a.(v + rv)
// All head rows are 16 contiguous halfs -> 8-half vector loads.
// ---------------------------------------------------------------------------
__global__ void k_attn(const _Float16* __restrict__ q, const _Float16* __restrict__ k,
                       const _Float16* __restrict__ v, const _Float16* __restrict__ rk,
                       const _Float16* __restrict__ rv, _Float16* __restrict__ o,
                       int L, int H)
{
    int i = blockIdx.x, h = blockIdx.y, b = blockIdx.z;
    int lane = threadIdx.x;
    int D = H * 16;
    __shared__ float sl[64];
    size_t rowQ = ((size_t)b * L + i) * D + h * 16;
    h8 q0 = *(const h8*)(q + rowQ);
    h8 q1 = *(const h8*)(q + rowQ + 8);
    float qv[16];
#pragma unroll
    for (int d = 0; d < 8; ++d) { qv[d] = (float)q0[d]; qv[d + 8] = (float)q1[d]; }
    for (int j = lane; j < L; j += 32) {
        size_t rowK = ((size_t)b * L + j) * D + h * 16;
        size_t rowR = (((size_t)b * L + i) * L + j) * D + h * 16;
        h8 k0 = *(const h8*)(k + rowK);
        h8 k1 = *(const h8*)(k + rowK + 8);
        h8 r0 = *(const h8*)(rk + rowR);
        h8 r1 = *(const h8*)(rk + rowR + 8);
        float s = 0.f;
#pragma unroll
        for (int d = 0; d < 8; ++d) {
            s += qv[d]     * ((float)k0[d] + (float)r0[d]);
            s += qv[d + 8] * ((float)k1[d] + (float)r1[d]);
        }
        sl[j] = s * 0.25f;
    }
    __syncthreads();
    float mx = -1e30f;
    for (int j = lane; j < L; j += 32) mx = fmaxf(mx, sl[j]);
#pragma unroll
    for (int off = 16; off; off >>= 1) mx = fmaxf(mx, __shfl_xor(mx, off, 32));
    float sum = 0.f;
    for (int j = lane; j < L; j += 32) { float e = expf(sl[j] - mx); sl[j] = e; sum += e; }
#pragma unroll
    for (int off = 16; off; off >>= 1) sum += __shfl_xor(sum, off, 32);
    __syncthreads();
    float inv = 1.f / sum;
    float acc[16];
#pragma unroll
    for (int d = 0; d < 16; ++d) acc[d] = 0.f;
    for (int j = lane; j < L; j += 32) {
        float a = sl[j] * inv;
        size_t rowV = ((size_t)b * L + j) * D + h * 16;
        size_t rowR = (((size_t)b * L + i) * L + j) * D + h * 16;
        h8 v0 = *(const h8*)(v + rowV);
        h8 v1 = *(const h8*)(v + rowV + 8);
        h8 r0 = *(const h8*)(rv + rowR);
        h8 r1 = *(const h8*)(rv + rowR + 8);
#pragma unroll
        for (int d = 0; d < 8; ++d) {
            acc[d]     += a * ((float)v0[d] + (float)r0[d]);
            acc[d + 8] += a * ((float)v1[d] + (float)r1[d]);
        }
    }
#pragma unroll
    for (int off = 16; off; off >>= 1) {
#pragma unroll
        for (int d = 0; d < 16; ++d) acc[d] += __shfl_xor(acc[d], off, 32);
    }
    if (lane == 0) {
#pragma unroll
        for (int d = 0; d < 16; ++d) o[rowQ + d] = (_Float16)acc[d];
    }
}

// ---------------------------------------------------------------------------
// Downsample h: channel-last (192,L,D) -> stride-2 conv1d k=3 -> (192,L/2,2D)
// ---------------------------------------------------------------------------
__global__ void k_down_h(const float* __restrict__ h, const float* __restrict__ w,
                         float* __restrict__ out, int L, int D)
{
    int D2 = 2 * D, L2 = L / 2;
    int total = 192 * L2 * D2;
    int idx = blockIdx.x * blockDim.x + threadIdx.x;
    if (idx >= total) return;
    int o  = idx % D2;
    int t2 = (idx / D2) % L2;
    int b  = idx / (D2 * L2);
    float s = 0.f;
    for (int kk = 0; kk < 3; ++kk) {
        int t = 2 * t2 + kk - 1;
        if (t < 0 || t >= L) continue;
        const float* hr = h + ((size_t)b * L + t) * D;
        const float* wr = w + (size_t)o * D * 3 + kk;
        for (int c = 0; c < D; ++c) s += hr[c] * wr[c * 3];
    }
    out[idx] = s;
}

// ---------------------------------------------------------------------------
// Downsample RPE: (192,L,L,D) -> stride-2 3x3 conv (matching NCHW transpose) -> (192,L/2,L/2,2D)
// new[b,i,j,o] = sum_{kh,kw,c} old[b, 2i+kw-1, 2j+kh-1, c] * w[o,c,kh,kw]
// ---------------------------------------------------------------------------
__global__ void k_down_rpe(const float* __restrict__ rpe, const float* __restrict__ w,
                           float* __restrict__ out, int L, int D)
{
    int D2 = 2 * D, L2 = L / 2;
    int total = 192 * L2 * L2 * D2;
    int idx = blockIdx.x * blockDim.x + threadIdx.x;
    if (idx >= total) return;
    int o  = idx % D2;
    int j2 = (idx / D2) % L2;
    int i2 = (idx / (D2 * L2)) % L2;
    int b  = idx / (D2 * L2 * L2);
    float s = 0.f;
    for (int kh = 0; kh < 3; ++kh) {
        int jt = 2 * j2 + kh - 1;
        if (jt < 0 || jt >= L) continue;
        for (int kw = 0; kw < 3; ++kw) {
            int it = 2 * i2 + kw - 1;
            if (it < 0 || it >= L) continue;
            const float* rr = rpe + (((size_t)b * L + it) * L + jt) * D;
            const float* wr = w + ((size_t)o * D) * 9 + kh * 3 + kw;
            for (int c = 0; c < D; ++c) s += rr[c] * wr[c * 9];
        }
    }
    out[idx] = s;
}

// ---------------------------------------------------------------------------
// Lateral conv: channel-last input (192,L,Dc) * w(128,Dc,3)+b -> (192,128,L)
// ---------------------------------------------------------------------------
__global__ void k_lateral(const float* __restrict__ in, const float* __restrict__ w,
                          const float* __restrict__ bias, float* __restrict__ out,
                          int L, int Dc)
{
    int total = 192 * 128 * L;
    int idx = blockIdx.x * blockDim.x + threadIdx.x;
    if (idx >= total) return;
    int t = idx % L;
    int o = (idx / L) % 128;
    int b = idx / (128 * L);
    float s = bias[o];
    for (int kk = 0; kk < 3; ++kk) {
        int tt = t + kk - 1;
        if (tt < 0 || tt >= L) continue;
        const float* ir = in + ((size_t)b * L + tt) * Dc;
        const float* wr = w + (size_t)o * Dc * 3 + kk;
        for (int c = 0; c < Dc; ++c) s += ir[c] * wr[c * 3];
    }
    out[idx] = s;
}

// ---------------------------------------------------------------------------
// Linear upsample (2x) + add:  out(192,128,Lout) += upsample(in(192,128,Lin))
// ---------------------------------------------------------------------------
__global__ void k_upsample_add(float* __restrict__ out, const float* __restrict__ in,
                               int Lout, int Lin)
{
    int total = 192 * 128 * Lout;
    int idx = blockIdx.x * blockDim.x + threadIdx.x;
    if (idx >= total) return;
    int t = idx % Lout;
    int r = idx / Lout;  // b*128 + o
    float src = ((float)t + 0.5f) * ((float)Lin / (float)Lout) - 0.5f;
    src = fminf(fmaxf(src, 0.f), (float)(Lin - 1));
    int i0 = (int)floorf(src);
    int i1 = min(i0 + 1, Lin - 1);
    float wgt = src - (float)i0;
    out[idx] += in[(size_t)r * Lin + i0] * (1.f - wgt) + in[(size_t)r * Lin + i1] * wgt;
}

// ---------------------------------------------------------------------------
// FPN head conv: channel-first in (192,128,64) * w(128,128,3)+b -> out (192,128,64)
// ---------------------------------------------------------------------------
__global__ void k_fpn(const float* __restrict__ in, const float* __restrict__ w,
                      const float* __restrict__ bias, float* __restrict__ out)
{
    int total = 192 * 128 * 64;
    int idx = blockIdx.x * blockDim.x + threadIdx.x;
    if (idx >= total) return;
    int t = idx & 63;
    int o = (idx >> 6) & 127;
    int b = idx >> 13;
    float s = bias[o];
    for (int c = 0; c < 128; ++c) {
        const float* ir = in + ((size_t)b * 128 + c) * 64;
        const float* wr = w + ((size_t)o * 128 + c) * 3;
#pragma unroll
        for (int kk = 0; kk < 3; ++kk) {
            int tt = t + kk - 1;
            if (tt >= 0 && tt < 64) s += ir[tt] * wr[kk];
        }
    }
    out[idx] = s;
}

// ===========================================================================
// Host orchestration
// ===========================================================================
extern "C" void kernel_launch(void* const* d_in, const int* in_sizes, int n_in,
                              void* d_out, int out_size, void* d_ws, size_t ws_size,
                              hipStream_t stream)
{
    (void)in_sizes; (void)out_size; (void)ws_size;
    if (n_in < 127) return;
    auto F = [&](int i) { return (const float*)d_in[i]; };
    const float* x = F(0);

    // JAX pytree (sorted dict keys) leaf indices
    const int I_EMBED_B = 1, I_EMBED_W = 2, I_FPN_B = 3, I_FPN_W = 4;
    const int I_LAT_B0 = 5, I_LAT_W0 = 8;
    const int I_RTW = 125, I_RTB = 126;
    const int I_ONG0 = 119;  // out_norms: g0,b0,g1,b1,g2,b2
    // per-(level,block) base: leaves sorted W1,W2,Wk,Wo,Wq,Wv,b1,b2,bk,bo,bq,bv,n1b,n1g,n2b,n2g
    const int BLK[3][2] = {{11, 27}, {49, 65}, {87, 103}};
    const int DOWN[2]   = {43, 81};  // down_b, down_g(+1), down_w(+2), rped_b(+3), rped_g(+4), rped_w(+5)

    // Workspace arena
    char* base = (char*)d_ws;
    size_t off = 0;
    auto ALLOCF = [&](size_t n) { float* p = (float*)(base + off);
                                  off += ((n * 4 + 255) / 256) * 256; return p; };
    auto ALLOCH = [&](size_t n) { _Float16* p = (_Float16*)(base + off);
                                  off += ((n * 2 + 255) / 256) * 256; return p; };
    const size_t RPE0 = (size_t)192 * 64 * 64 * 32;
    const size_t RPE1 = (size_t)192 * 32 * 32 * 64;
    const size_t HSZ  = (size_t)192 * 64 * 32;
    float*    rpeA   = ALLOCF(RPE0);
    float*    rpeB   = ALLOCF(RPE1);
    _Float16* rpe16  = ALLOCH(RPE0);
    _Float16* rk16   = ALLOCH(RPE0);
    _Float16* rv16   = ALLOCH(RPE0);
    float*    h      = ALLOCF(HSZ);
    float*    y32    = ALLOCF(HSZ);
    float*    hn32   = ALLOCF(HSZ);
    float*    hnext  = ALLOCF(HSZ);
    _Float16* hn16   = ALLOCH(HSZ);
    _Float16* h216   = ALLOCH(HSZ);
    _Float16* q16    = ALLOCH(HSZ);
    _Float16* kk16   = ALLOCH(HSZ);
    _Float16* vv16   = ALLOCH(HSZ);
    _Float16* o16    = ALLOCH(HSZ);
    _Float16* mlph16 = ALLOCH(3 * HSZ);
    _Float16* wq16   = ALLOCH(16384);
    _Float16* wk16   = ALLOCH(16384);
    _Float16* wv16   = ALLOCH(16384);
    _Float16* wo16   = ALLOCH(16384);
    _Float16* w116   = ALLOCH(49152);
    _Float16* w216   = ALLOCH(49152);
    float* lat[3];
    lat[0] = ALLOCF((size_t)192 * 128 * 64);
    lat[1] = ALLOCF((size_t)192 * 128 * 32);
    lat[2] = ALLOCF((size_t)192 * 128 * 16);

    auto stage = [&](const float* Win, _Float16* outp, int K, int N) {
        int n = K * N;
        k_stage_B<<<(n + 255) / 256, 256, 0, stream>>>(Win, outp, K, N);
    };
    auto ln = [&](const float* xin, const float* g, const float* b,
                  float* y, _Float16* yh, int M, int D) {
        k_layernorm<<<M, 32, 0, stream>>>(xin, g, b, y, yh, M, D);
    };
    auto gemm = [&](const _Float16* A, const _Float16* Bm, const float* bias,
                    const float* addend, float* C, _Float16* Ch,
                    int M, int N, int K, int act) {
        dim3 g(M / 64, N / 16, 1), blk(32, 4, 1);
        k_gemm_f16<<<g, blk, 0, stream>>>(A, Bm, bias, addend, C, Ch, M, N, K, act);
    };

    // RPE level-0 + embed
    {
        int total = 192 * 64 * 64;
        k_build_rpe<<<(total + 255) / 256, 256, 0, stream>>>(x, F(I_RTW), F(I_RTB), rpeA, rpe16);
    }
    {
        int total = 192 * 64 * 32;
        k_embed<<<(total + 255) / 256, 256, 0, stream>>>(x, F(I_EMBED_W), F(I_EMBED_B), h);
    }

    const int Ls[3] = {64, 32, 16}, Ds[3] = {32, 64, 128}, Hs[3] = {2, 4, 8};
    for (int lvl = 0; lvl < 3; ++lvl) {
        int L = Ls[lvl], D = Ds[lvl], H = Hs[lvl];
        int BL = 192 * L, BLL = 192 * L * L;
        for (int blk = 0; blk < 2; ++blk) {
            int pb = BLK[lvl][blk];
            stage(F(pb + 4), wq16, D, D);        // Wq
            stage(F(pb + 2), wk16, D, D);        // Wk
            stage(F(pb + 5), wv16, D, D);        // Wv
            stage(F(pb + 3), wo16, D, D);        // Wo
            stage(F(pb + 0), w116, D, 3 * D);    // W1
            stage(F(pb + 1), w216, 3 * D, D);    // W2
            // LN1
            ln(h, F(pb + 13), F(pb + 12), nullptr, hn16, BL, D);
            // Q/K/V projections (WMMA)
            gemm(hn16, wq16, F(pb + 10), nullptr, nullptr, q16, BL, D, D, 0);
            gemm(hn16, wk16, F(pb + 8),  nullptr, nullptr, kk16, BL, D, D, 0);
            gemm(hn16, wv16, F(pb + 11), nullptr, nullptr, vv16, BL, D, D, 0);
            // rk = rpe @ Wk, rv = rpe @ Wv (WMMA, M = B*L*L)
            gemm(rpe16, wk16, nullptr, nullptr, nullptr, rk16, BLL, D, D, 0);
            gemm(rpe16, wv16, nullptr, nullptr, nullptr, rv16, BLL, D, D, 0);
            // attention softmax
            k_attn<<<dim3(L, H, 192), 32, 0, stream>>>(q16, kk16, vv16, rk16, rv16, o16, L, H);
            // out-proj + residual (WMMA): y = s + o@Wo + bo
            gemm(o16, wo16, F(pb + 9), h, y32, nullptr, BL, D, D, 0);
            // LN2
            ln(y32, F(pb + 15), F(pb + 14), nullptr, h216, BL, D);
            // MLP (WMMA): gelu(h2@W1+b1) @ W2 + b2, residual to s
            gemm(h216, w116, F(pb + 6), nullptr, nullptr, mlph16, BL, 3 * D, D, 1);
            gemm(mlph16, w216, F(pb + 7), h, h, nullptr, BL, D, 3 * D, 0);
        }
        // out_norm + lateral conv
        ln(h, F(I_ONG0 + 2 * lvl), F(I_ONG0 + 2 * lvl + 1), hn32, nullptr, BL, D);
        {
            int total = 192 * 128 * L;
            k_lateral<<<(total + 255) / 256, 256, 0, stream>>>(
                hn32, F(I_LAT_W0 + lvl), F(I_LAT_B0 + lvl), lat[lvl], L, D);
        }
        if (lvl < 2) {
            int dn = DOWN[lvl];
            int L2 = L / 2, D2 = 2 * D;
            {
                int total = 192 * L2 * D2;
                k_down_h<<<(total + 255) / 256, 256, 0, stream>>>(h, F(dn + 2), hnext, L, D);
            }
            ln(hnext, F(dn + 1), F(dn + 0), h, nullptr, 192 * L2, D2);
            {
                int total = 192 * L2 * L2 * D2;
                k_down_rpe<<<(total + 255) / 256, 256, 0, stream>>>(rpeA, F(dn + 5), rpeB, L, D);
            }
            ln(rpeB, F(dn + 4), F(dn + 3), rpeA, rpe16, 192 * L2 * L2, D2);
        }
    }

    // FPN: lats[1] += up(lats[2]); lats[0] += up(lats[1]); out = conv(lats[0])
    {
        int total = 192 * 128 * 32;
        k_upsample_add<<<(total + 255) / 256, 256, 0, stream>>>(lat[1], lat[2], 32, 16);
    }
    {
        int total = 192 * 128 * 64;
        k_upsample_add<<<(total + 255) / 256, 256, 0, stream>>>(lat[0], lat[1], 64, 32);
    }
    {
        int total = 192 * 128 * 64;
        k_fpn<<<(total + 255) / 256, 256, 0, stream>>>(lat[0], F(I_FPN_W), F(I_FPN_B), (float*)d_out);
    }
}
